// ResidualQuantize_19859928777279
// MI455X (gfx1250) — matrix operand order repaired and checked by reading, not compiled, not run on hardware
//
#include <hip/hip_runtime.h>
#include <math.h>

typedef __attribute__((ext_vector_type(16))) _Float16 v16h;
typedef __attribute__((ext_vector_type(8)))  _Float16 v8h;
typedef __attribute__((ext_vector_type(4)))  _Float16 v4h;
typedef __attribute__((ext_vector_type(8)))  float    v8f;
typedef __attribute__((ext_vector_type(4)))  unsigned int uint4v;
typedef __attribute__((ext_vector_type(8)))  int      int8v;
typedef __attribute__((ext_vector_type(4)))  int      int4v;

#define BS_ROWS 8192      // B*S
#define DIM     768
#define NCODE   1024
#define HID     512
#define DHID    1536

// ---------------------------------------------------------------------------
// init: residual = x ; qout = 0
// ---------------------------------------------------------------------------
__global__ __launch_bounds__(256) void rvq_init_kernel(
    const float* __restrict__ x, float* __restrict__ residual,
    float* __restrict__ qout, int n) {
  int i = blockIdx.x * 256 + threadIdx.x;
  if (i < n) {
    residual[i] = x[i];
    qout[i] = 0.0f;
  }
}

// ---------------------------------------------------------------------------
// Weight prep (once per launch): W[K][N] f32 -> Wt[N][K] f16 via 32x32 LDS
// transpose tiles so both global streams are coalesced. K,N % 32 == 0.
// ---------------------------------------------------------------------------
__global__ __launch_bounds__(256) void rvq_wt_f16_kernel(
    const float* __restrict__ W, _Float16* __restrict__ Wt, int K, int N) {
  __shared__ float t[32][33];
  const int j  = threadIdx.x & 31;
  const int i0 = threadIdx.x >> 5;   // 0..7
  const int nbase = blockIdx.x * 32;
  const int kbase = blockIdx.y * 32;
#pragma unroll
  for (int i = i0; i < 32; i += 8)
    t[i][j] = W[(size_t)(kbase + i) * N + nbase + j];
  __syncthreads();
#pragma unroll
  for (int i = i0; i < 32; i += 8)
    Wt[(size_t)(nbase + i) * K + kbase + j] = (_Float16)t[j][i];
}

// ---------------------------------------------------------------------------
// WMMA GEMM with fused bias + eval-BatchNorm + ReLU epilogue.
//   Out[M,N] = relu( bn( A[M,K] @ W[K,N] + bias ) ),  W given as Wt[N][K] f16.
// Block tile 128x128, 256 threads (8 wave32 in 4x2), wave tile 32x64,
// K stepped by 32 using v_wmma_f32_16x16x32_f16.
// B tile (128 n-rows x 32 k f16, 64B/row + 16B pad -> 80B LDS stride) is
// DMA'd by the Tensor Data Mover: one tensor_load_to_lds per K-step issued by
// wave 0 (uniform SGPR descriptor), completed with s_wait_tensorcnt + barrier.
// A tile: 4 independent global_load_b128 (one clause / one wait), then
// v_cvt_pk_f16_f32 + ds_store_b64.
// M % 128 == 0, N % 128 == 0, K % 32 == 0 (true for all call sites).
// ---------------------------------------------------------------------------
__global__ __launch_bounds__(256) void rvq_gemm_bn_relu_kernel(
    const float* __restrict__ A, const _Float16* __restrict__ Wt,
    const float* __restrict__ bias, const float* __restrict__ gamma,
    const float* __restrict__ bnbeta, const float* __restrict__ mean,
    const float* __restrict__ var, float* __restrict__ Out,
    int N, int K) {
  __shared__ alignas(16) _Float16 As[128][40];  // [m][k], stride 40 halves
  __shared__ alignas(16) _Float16 Bs[128][40];  // [n][k], TDM-filled

  const int tid   = threadIdx.x;
  const int lane  = tid & 31;
  const int wave  = tid >> 5;
  const int waveM = wave & 3;    // 0..3  -> 32-row strip
  const int waveN = wave >> 2;   // 0..1  -> 64-col strip
  const int grp   = lane >> 4;   // 0 or 1 (half-wave)
  const int l16   = lane & 15;
  const int rowBase = blockIdx.x * 128;
  const int colBase = blockIdx.y * 128;
  const unsigned int ldsBs = (unsigned int)(uintptr_t)&Bs[0][0];

  // per-thread A-tile slots (constant across K steps)
  int ar[4], ac4[4];
#pragma unroll
  for (int it = 0; it < 4; ++it) {
    int idx = tid + it * 256;           // over 128 rows x 8 float4
    ar[it]  = idx >> 3;
    ac4[it] = idx & 7;
  }

  v8f acc[2][4] = {};

  for (int kk = 0; kk < K; kk += 32) {
    // --- A tile 128x32 f32: batch the 4 global loads, then convert+store ---
    float4 f[4];
#pragma unroll
    for (int it = 0; it < 4; ++it)
      f[it] = *(const float4*)(A + (size_t)(rowBase + ar[it]) * K + kk +
                               ac4[it] * 4);
#pragma unroll
    for (int it = 0; it < 4; ++it) {
      v4h hv;
      hv[0] = (_Float16)f[it].x; hv[1] = (_Float16)f[it].y;
      hv[2] = (_Float16)f[it].z; hv[3] = (_Float16)f[it].w;
      *(v4h*)&As[ar[it]][ac4[it] * 4] = hv;
    }

    // --- B tile via Tensor Data Mover: Wt[colBase..+127][kk..+31] -> Bs ---
    if (tid < 32) {  // wave 0 issues the DMA (EXEC ignored by TDM)
      unsigned long long ga =
          (unsigned long long)(uintptr_t)(Wt + (size_t)colBase * K + kk);
      // D# group 0: count=1 | lds_addr | global_addr | type=2
      uint4v g0;
      g0[0] = 1u;                                   // count = 1 descriptor
      g0[1] = ldsBs;                                // LDS byte address
      g0[2] = (unsigned int)ga;                     // global_addr[31:0]
      g0[3] = ((unsigned int)(ga >> 32) & 0x01FFFFFFu) | 0x80000000u; // type=2
      // D# group 1: data_size=2B, pad 4 DWORDs every 16 DWORDs,
      // tensor_dim0=K, tensor_dim1=N, tile_dim0=32, tile_dim1=128,
      // tensor_dim0_stride=K
      int8v g1;
      g1[0] = (int)((1u << 16) | (1u << 20) | (3u << 22) | (3u << 25));
      g1[1] = (int)(((unsigned int)K & 0xFFFFu) << 16);          // dim0 lo16
      g1[2] = (int)(((unsigned int)K >> 16) |
                    (((unsigned int)N & 0xFFFFu) << 16));        // dim0 hi/dim1 lo
      g1[3] = (int)(((unsigned int)N >> 16) | (32u << 16));      // dim1 hi | tile_dim0
      g1[4] = 128;                                               // tile_dim1
      g1[5] = K;                                                 // dim0_stride lo32
      g1[6] = 0;
      g1[7] = 0;
      int4v gz4 = {0, 0, 0, 0};
      int8v gz8 = {0, 0, 0, 0, 0, 0, 0, 0};
      __builtin_amdgcn_tensor_load_to_lds(g0, g1, gz4, gz4, gz8, 0);
      __builtin_amdgcn_s_wait_tensorcnt(0);
    }
    __syncthreads();

    // --- fragments (ISA 7.12.2 16-bit A/B layouts) ---
    v16h afrag[2], bfrag[4];
#pragma unroll
    for (int mt = 0; mt < 2; ++mt) {
      int m = waveM * 32 + mt * 16 + l16;
      v8h lo = *(const v8h*)&As[m][grp * 8];        // k = grp*8 + 0..7
      v8h hi = *(const v8h*)&As[m][16 + grp * 8];   // k = 16 + grp*8 + 0..7
      afrag[mt] = __builtin_shufflevector(lo, hi, 0, 1, 2, 3, 4, 5, 6, 7, 8, 9,
                                          10, 11, 12, 13, 14, 15);
    }
#pragma unroll
    for (int nt = 0; nt < 4; ++nt) {
      int n = waveN * 64 + nt * 16 + l16;
      v8h lo = *(const v8h*)&Bs[n][grp * 8];        // k = grp*8 + 0..7
      v8h hi = *(const v8h*)&Bs[n][16 + grp * 8];   // k = 16 + grp*8 + 0..7
      bfrag[nt] = __builtin_shufflevector(lo, hi, 0, 1, 2, 3, 4, 5, 6, 7, 8, 9,
                                          10, 11, 12, 13, 14, 15);
    }
#pragma unroll
    for (int mt = 0; mt < 2; ++mt)
#pragma unroll
      for (int nt = 0; nt < 4; ++nt)
        acc[mt][nt] = __builtin_amdgcn_wmma_f32_16x16x32_f16(
            false, afrag[mt], false, bfrag[nt], (short)0, acc[mt][nt], false,
            false);
    __syncthreads();
  }

  // --- epilogue: y = relu( (t + bias - mean)*g*rsqrt(v+eps) + beta ) ---
#pragma unroll
  for (int nt = 0; nt < 4; ++nt) {
    int col = colBase + waveN * 64 + nt * 16 + l16;
    float s  = gamma[col] * rsqrtf(var[col] + 1e-5f);
    float sh = (bias[col] - mean[col]) * s + bnbeta[col];
#pragma unroll
    for (int mt = 0; mt < 2; ++mt) {
#pragma unroll
      for (int r = 0; r < 8; ++r) {
        int row = rowBase + waveM * 32 + mt * 16 + grp * 8 + r;
        float t = acc[mt][nt][r] * s + sh;
        Out[(size_t)row * N + col] = t > 0.0f ? t : 0.0f;
      }
    }
  }
}

// ---------------------------------------------------------------------------
// Per-row quantizer tail: softmax stats, entropy, latent loss, gumbel argmax,
// codebook gather, residual/qout update. One block (256 thr) per token row.
// y_hard + y_soft - sg(y_soft) == y_hard numerically, so
// z_q = codebook[argmax(z + gumbel)]  (TAU > 0 doesn't change the argmax).
// ---------------------------------------------------------------------------
__global__ __launch_bounds__(256) void rvq_rowquant_kernel(
    const float* __restrict__ z, const float* __restrict__ gumbel,
    const float* __restrict__ codebook, float* __restrict__ entropy_out,
    float* __restrict__ latent_out, float* __restrict__ idx_out,
    float* __restrict__ residual, float* __restrict__ qout) {
  const int row = blockIdx.x;
  const int tid = threadIdx.x;
  __shared__ float red[256];
  __shared__ int   redi[256];

  const float* zr = z + (size_t)row * NCODE;
  const float* gr = gumbel + (size_t)row * NCODE;

  float zv[4];
  float zmax = -3.4e38f, lmax = -3.4e38f;
  int lidx = 0;
#pragma unroll
  for (int j = 0; j < 4; ++j) {
    int n = tid + j * 256;
    zv[j] = zr[n];
    float lg = zv[j] + gr[n];
    if (zv[j] > zmax) zmax = zv[j];
    if (lg > lmax) { lmax = lg; lidx = n; }
  }

  // reduce row max of z
  red[tid] = zmax; __syncthreads();
  for (int s = 128; s > 0; s >>= 1) {
    if (tid < s) red[tid] = fmaxf(red[tid], red[tid + s]);
    __syncthreads();
  }
  float rmax = red[0]; __syncthreads();

  // reduce sum of exp
  float ev[4], se = 0.0f;
#pragma unroll
  for (int j = 0; j < 4; ++j) { ev[j] = expf(zv[j] - rmax); se += ev[j]; }
  red[tid] = se; __syncthreads();
  for (int s = 128; s > 0; s >>= 1) {
    if (tid < s) red[tid] += red[tid + s];
    __syncthreads();
  }
  float inv = 1.0f / red[0]; __syncthreads();

  // entropy + latent loss
  float ent = 0.0f;
#pragma unroll
  for (int j = 0; j < 4; ++j) {
    int n = tid + j * 256;
    float q = ev[j] * inv;
    ent -= q * logf(q + 1e-10f);
    latent_out[(size_t)row * NCODE + n] = q * logf(q * (float)NCODE + 1e-10f);
  }
  red[tid] = ent; __syncthreads();
  for (int s = 128; s > 0; s >>= 1) {
    if (tid < s) red[tid] += red[tid + s];
    __syncthreads();
  }
  if (tid == 0) entropy_out[row] = red[0];
  __syncthreads();

  // argmax of z + gumbel (first-occurrence tie break)
  red[tid] = lmax; redi[tid] = lidx; __syncthreads();
  for (int s = 128; s > 0; s >>= 1) {
    if (tid < s) {
      float ov = red[tid + s]; int oi = redi[tid + s];
      if (ov > red[tid] || (ov == red[tid] && oi < redi[tid])) {
        red[tid] = ov; redi[tid] = oi;
      }
    }
    __syncthreads();
  }
  int idx = redi[0];
  if (tid == 0) idx_out[row] = (float)idx;

  // z_q = codebook[idx]; residual -= z_q; qout += z_q
  const float* cb = codebook + (size_t)idx * DIM;
  float* rr = residual + (size_t)row * DIM;
  float* qq = qout + (size_t)row * DIM;
  for (int d = tid; d < DIM; d += 256) {
    float c = cb[d];
    rr[d] -= c;
    qq[d] += c;
  }
}

// ---------------------------------------------------------------------------
extern "C" void kernel_launch(void* const* d_in, const int* in_sizes, int n_in,
                              void* d_out, int out_size, void* d_ws,
                              size_t ws_size, hipStream_t stream) {
  const float* x    = (const float*)d_in[0];
  const float* cbks = (const float*)d_in[1];
  const float* pW1  = (const float*)d_in[2];
  const float* pb1  = (const float*)d_in[3];
  const float* pg1  = (const float*)d_in[4];
  const float* pbe1 = (const float*)d_in[5];
  const float* pm1  = (const float*)d_in[6];
  const float* pv1  = (const float*)d_in[7];
  const float* pW2  = (const float*)d_in[8];
  const float* pb2  = (const float*)d_in[9];
  const float* pg2  = (const float*)d_in[10];
  const float* pbe2 = (const float*)d_in[11];
  const float* pm2  = (const float*)d_in[12];
  const float* pv2  = (const float*)d_in[13];
  const float* dW1  = (const float*)d_in[14];
  const float* db1  = (const float*)d_in[15];
  const float* dg1  = (const float*)d_in[16];
  const float* dbe1 = (const float*)d_in[17];
  const float* dm1  = (const float*)d_in[18];
  const float* dv1  = (const float*)d_in[19];
  const float* dW2  = (const float*)d_in[20];
  const float* db2  = (const float*)d_in[21];
  const float* dg2  = (const float*)d_in[22];
  const float* dbe2 = (const float*)d_in[23];
  const float* dm2  = (const float*)d_in[24];
  const float* dv2  = (const float*)d_in[25];
  const float* gmb  = (const float*)d_in[26];

  // outputs concatenated: xhat, entropies, latent, indices, zs
  float* out  = (float*)d_out;
  float* xhat = out;
  float* ent  = xhat + (size_t)BS_ROWS * DIM;
  float* lat  = ent + 4 * (size_t)BS_ROWS;
  float* idxo = lat + 4 * (size_t)BS_ROWS * NCODE;
  float* zs   = idxo + 4 * (size_t)BS_ROWS;

  // workspace: residual[8192x768], qout[8192x768], h[8192x1536], f16 weights
  float* residual = (float*)d_ws;
  float* qout     = residual + (size_t)BS_ROWS * DIM;
  float* h        = qout + (size_t)BS_ROWS * DIM;
  _Float16* pW1t  = (_Float16*)(h + (size_t)BS_ROWS * DHID);  // 4 x [512][768]
  _Float16* pW2t  = pW1t + 4 * (size_t)HID * DIM;             // 4 x [1024][512]
  _Float16* dW1t  = pW2t + 4 * (size_t)NCODE * HID;           // [1536][768]
  _Float16* dW2t  = dW1t + (size_t)DHID * DIM;                // [768][1536]

  // once-per-launch weight convert+transpose to f16 Wt[N][K]
  for (int i = 0; i < 4; ++i) {
    rvq_wt_f16_kernel<<<dim3(HID / 32, DIM / 32), 256, 0, stream>>>(
        pW1 + (size_t)i * DIM * HID, pW1t + (size_t)i * HID * DIM, DIM, HID);
    rvq_wt_f16_kernel<<<dim3(NCODE / 32, HID / 32), 256, 0, stream>>>(
        pW2 + (size_t)i * HID * NCODE, pW2t + (size_t)i * NCODE * HID, HID,
        NCODE);
  }
  rvq_wt_f16_kernel<<<dim3(DHID / 32, DIM / 32), 256, 0, stream>>>(
      dW1, dW1t, DIM, DHID);
  rvq_wt_f16_kernel<<<dim3(DIM / 32, DHID / 32), 256, 0, stream>>>(
      dW2, dW2t, DHID, DIM);

  rvq_init_kernel<<<(BS_ROWS * DIM + 255) / 256, 256, 0, stream>>>(
      x, residual, qout, BS_ROWS * DIM);

  for (int i = 0; i < 4; ++i) {
    // h = blk(residual, pW1[i], ...)   [8192,768] x [768,512]
    rvq_gemm_bn_relu_kernel<<<dim3(BS_ROWS / 128, HID / 128), 256, 0, stream>>>(
        residual, pW1t + (size_t)i * HID * DIM, pb1 + i * HID, pg1 + i * HID,
        pbe1 + i * HID, pm1 + i * HID, pv1 + i * HID, h, HID, DIM);
    // z = blk(h, pW2[i], ...)          [8192,512] x [512,1024] -> zs[i]
    float* zi = zs + (size_t)i * BS_ROWS * NCODE;
    rvq_gemm_bn_relu_kernel<<<dim3(BS_ROWS / 128, NCODE / 128), 256, 0,
                              stream>>>(
        h, pW2t + (size_t)i * NCODE * HID, pb2 + i * NCODE, pg2 + i * NCODE,
        pbe2 + i * NCODE, pm2 + i * NCODE, pv2 + i * NCODE, zi, NCODE, HID);
    // softmax stats + gumbel argmax + codebook gather + residual update
    rvq_rowquant_kernel<<<BS_ROWS, 256, 0, stream>>>(
        zi, gmb + (size_t)i * BS_ROWS * NCODE, cbks + (size_t)i * NCODE * DIM,
        ent + (size_t)i * BS_ROWS, lat + (size_t)i * BS_ROWS * NCODE,
        idxo + (size_t)i * BS_ROWS, residual, qout);
  }

  // decoder: h = blk(qout, dW1...), xhat = blk(h, dW2...)
  rvq_gemm_bn_relu_kernel<<<dim3(BS_ROWS / 128, DHID / 128), 256, 0, stream>>>(
      qout, dW1t, db1, dg1, dbe1, dm1, dv1, h, DHID, DIM);
  rvq_gemm_bn_relu_kernel<<<dim3(BS_ROWS / 128, DIM / 128), 256, 0, stream>>>(
      h, dW2t, db2, dg2, dbe2, dm2, dv2, xhat, DIM, DHID);
}
